// Stochastic_Linear_Synapse_42545946034246
// MI455X (gfx1250) — matrix operand order repaired and checked by reading, not compiled, not run on hardware
//
#include <hip/hip_runtime.h>
#include <stdint.h>

#define B_DIM   128
#define OUT_DIM 1024
#define IN_DIM  1024

typedef __attribute__((ext_vector_type(16))) __bf16 v16bf;
typedef __attribute__((ext_vector_type(8)))  float  v8f;

union Frag {
    uint4    u4[2];
    uint32_t u[8];
    v16bf    v;
};

__device__ inline uint32_t f32_to_bf16_bits(float f) {
    uint32_t x = __float_as_uint(f);
    // round-to-nearest-even truncation to bf16
    uint32_t r = x + 0x7fffu + ((x >> 16) & 1u);
    return r >> 16;
}

__device__ inline uint32_t pack_bf16x2(float lo, float hi) {
    return f32_to_bf16_bits(lo) | (f32_to_bf16_bits(hi) << 16);
}

// ---------------------------------------------------------------------------
// Kernel 1: s[o,i] = d + a*sigmoid(dx*(w - x0)); emit fp32 (for Bernoulli
// compare) + packed bf16 s and s^2 (for WMMA). One thread = 2 elements so we
// can write one packed dword per array.
// ---------------------------------------------------------------------------
__global__ __launch_bounds__(256) void k_prep_s(
    const float* __restrict__ w,  const float* __restrict__ x0,
    const float* __restrict__ dx, const float* __restrict__ a,
    const float* __restrict__ d,
    float* __restrict__ s_f32, uint32_t* __restrict__ s_bf,
    uint32_t* __restrict__ s2_bf)
{
    int i  = blockIdx.x * blockDim.x + threadIdx.x;   // over OUT*IN/2
    int e0 = 2 * i, e1 = e0 + 1;

    float t0 = dx[e0] * (w[e0] - x0[e0]);
    float t1 = dx[e1] * (w[e1] - x0[e1]);
    float g0 = 1.0f / (1.0f + __expf(-t0));
    float g1 = 1.0f / (1.0f + __expf(-t1));
    float s0 = d[e0] + a[e0] * g0;
    float s1 = d[e1] + a[e1] * g1;

    s_f32[e0] = s0;
    s_f32[e1] = s1;
    s_bf[i]   = pack_bf16x2(s0, s1);
    s2_bf[i]  = pack_bf16x2(s0 * s0, s1 * s1);
}

// ---------------------------------------------------------------------------
// Kernel 2: pack input and input^2 to bf16 pairs.
// ---------------------------------------------------------------------------
__global__ __launch_bounds__(256) void k_prep_in(
    const float* __restrict__ input,
    uint32_t* __restrict__ in_bf, uint32_t* __restrict__ in2_bf)
{
    int i  = blockIdx.x * blockDim.x + threadIdx.x;   // over B*IN/2
    int e0 = 2 * i, e1 = e0 + 1;
    float v0 = input[e0], v1 = input[e1];
    in_bf[i]  = pack_bf16x2(v0, v1);
    in2_bf[i] = pack_bf16x2(v0 * v0, v1 * v1);
}

// ---------------------------------------------------------------------------
// Kernel 3: mu[b,o] = sum_i in*s  and  q[b,o] = sum_i in^2*s^2 via
// v_wmma_f32_16x16x32_bf16. One wave per 16x16 output tile, 4 waves/block.
// Fragment gathers are two b128 loads per fragment per k-step (L2-resident).
// ---------------------------------------------------------------------------
__global__ __launch_bounds__(128) void k_wmma_mu_q(
    const uint32_t* __restrict__ in_bf, const uint32_t* __restrict__ in2_bf,
    const uint32_t* __restrict__ s_bf,  const uint32_t* __restrict__ s2_bf,
    float* __restrict__ mu, float* __restrict__ q)
{
    const int lane = threadIdx.x & 31;
    const int wave = threadIdx.x >> 5;
    const int tile = blockIdx.x * 4 + wave;        // 512 tiles total
    const int tm   = tile & 7;                     // B/16  = 8
    const int tn   = tile >> 3;                    // OUT/16 = 64

    const int row  = lane & 15;
    const int half = lane >> 4;
    const int rd   = IN_DIM / 2;                   // dwords per row

    const uint32_t* aRow  = in_bf  + (tm * 16 + row) * rd;
    const uint32_t* a2Row = in2_bf + (tm * 16 + row) * rd;
    const uint32_t* bRow  = s_bf   + (tn * 16 + row) * rd;
    const uint32_t* b2Row = s2_bf  + (tn * 16 + row) * rd;

    // A-matrix 16x32 bf16 layout: v0..3 -> dwords 4*half.., v4..7 -> 8+4*half..
    const int ja0 = 4 * half, ja1 = 8 + 4 * half;
    // B-matrix 32x16 layout (same K order as A): v0..3 -> dwords 4*half.., v4..7 -> 8+4*half..
    const int jb0 = 4 * half, jb1 = 8 + 4 * half;

    v8f accMu = {};
    v8f accQ  = {};

    for (int kt = 0; kt < IN_DIM / 32; ++kt) {
        const int base = kt * 16;                  // 16 dwords = 32 bf16 per row
        Frag A, A2, Bf, B2;
        A.u4[0]  = *(const uint4*)(aRow  + base + ja0);
        A.u4[1]  = *(const uint4*)(aRow  + base + ja1);
        A2.u4[0] = *(const uint4*)(a2Row + base + ja0);
        A2.u4[1] = *(const uint4*)(a2Row + base + ja1);
        Bf.u4[0] = *(const uint4*)(bRow  + base + jb0);
        Bf.u4[1] = *(const uint4*)(bRow  + base + jb1);
        B2.u4[0] = *(const uint4*)(b2Row + base + jb0);
        B2.u4[1] = *(const uint4*)(b2Row + base + jb1);

        accMu = __builtin_amdgcn_wmma_f32_16x16x32_bf16(
            false, A.v,  false, Bf.v, (short)0, accMu, false, false);
        accQ  = __builtin_amdgcn_wmma_f32_16x16x32_bf16(
            false, A2.v, false, B2.v, (short)0, accQ,  false, false);
    }

    // C/D layout: N = lane&15, M = v + 8*half
    const int n = tn * 16 + row;
#pragma unroll
    for (int v = 0; v < 8; ++v) {
        const int m = tm * 16 + v + 8 * half;
        mu[m * OUT_DIM + n] = accMu[v];
        q[m * OUT_DIM + n]  = accQ[v];
    }
}

// ---------------------------------------------------------------------------
// Kernel 4: Bernoulli counting (counter-based hash RNG) + final combine.
// Block owns a 16x16 (b,o) tile; input/s K-slabs staged through LDS.
// ---------------------------------------------------------------------------
__global__ __launch_bounds__(256) void k_bern_combine(
    const float* __restrict__ input, const float* __restrict__ s_f32,
    const float* __restrict__ mu,    const float* __restrict__ q,
    const float* __restrict__ bias,  float* __restrict__ out)
{
    __shared__ float lin[16][32];
    __shared__ float lsv[16][32];

    const int tx = threadIdx.x & 15;          // o within tile
    const int ty = threadIdx.x >> 4;          // b within tile
    const int bo = (blockIdx.x & 63) << 4;    // OUT/16 = 64 tiles
    const int bb = (blockIdx.x >> 6) << 4;    // B/16   = 8  tiles
    const int b  = bb + ty;
    const int o  = bo + tx;

    uint32_t cnt = 0;
    for (int k0 = 0; k0 < IN_DIM; k0 += 32) {
        const int t = threadIdx.x;
#pragma unroll
        for (int i = t; i < 512; i += 256) {
            const int r = i >> 5, c = i & 31;
            lin[r][c] = input[(bb + r) * IN_DIM + k0 + c];
            lsv[r][c] = s_f32[(bo + r) * IN_DIM + k0 + c];
        }
        __syncthreads();

        const uint32_t idxBase =
            (uint32_t)(b * OUT_DIM + o) * (uint32_t)IN_DIM + (uint32_t)k0;
#pragma unroll 4
        for (int kk = 0; kk < 32; ++kk) {
            float p = lin[ty][kk] * lsv[tx][kk];
            p = fminf(fmaxf(p, 0.0f), 1.0f);
            uint32_t h = idxBase + (uint32_t)kk;
            h ^= 0x9E3779B9u; h ^= h >> 16; h *= 0x7feb352du;
            h ^= h >> 15;     h *= 0x846ca68bu; h ^= h >> 16;
            float u = (float)(h >> 8) * (1.0f / 16777216.0f);
            cnt += (u < p) ? 1u : 0u;
        }
        __syncthreads();
    }

    const float Yb  = (float)cnt;
    const float m   = mu[b * OUT_DIM + o];
    const float qq  = q[b * OUT_DIM + o];
    const float var = fmaxf(m - qq, 0.0f);          // sum p(1-p)
    const float sig = sqrtf(var + 1e-4f);
    const float eps = (Yb - m) / sig;               // detached noise
    out[b * OUT_DIM + o] = m + eps * sig + bias[o];
}

// ---------------------------------------------------------------------------
// Launcher. Workspace layout (bytes):
//   s_f32 : OUT*IN*4          = 4 MB
//   s_bf  : OUT*IN/2 dwords   = 2 MB
//   s2_bf : OUT*IN/2 dwords   = 2 MB
//   in_bf : B*IN/2 dwords     = 256 KB
//   in2_bf: B*IN/2 dwords     = 256 KB
//   mu    : B*OUT*4           = 512 KB
//   q     : B*OUT*4           = 512 KB
// ---------------------------------------------------------------------------
extern "C" void kernel_launch(void* const* d_in, const int* in_sizes, int n_in,
                              void* d_out, int out_size, void* d_ws, size_t ws_size,
                              hipStream_t stream) {
    (void)in_sizes; (void)n_in; (void)out_size; (void)ws_size;

    const float* input  = (const float*)d_in[0];
    const float* weight = (const float*)d_in[1];
    const float* bias   = (const float*)d_in[2];
    const float* x0     = (const float*)d_in[3];
    const float* dx     = (const float*)d_in[4];
    const float* a      = (const float*)d_in[5];
    const float* dpar   = (const float*)d_in[6];
    float* out          = (float*)d_out;

    char* ws = (char*)d_ws;
    float*    s_f32  = (float*)ws;     ws += (size_t)OUT_DIM * IN_DIM * 4;
    uint32_t* s_bf   = (uint32_t*)ws;  ws += (size_t)OUT_DIM * IN_DIM / 2 * 4;
    uint32_t* s2_bf  = (uint32_t*)ws;  ws += (size_t)OUT_DIM * IN_DIM / 2 * 4;
    uint32_t* in_bf  = (uint32_t*)ws;  ws += (size_t)B_DIM * IN_DIM / 2 * 4;
    uint32_t* in2_bf = (uint32_t*)ws;  ws += (size_t)B_DIM * IN_DIM / 2 * 4;
    float*    mu     = (float*)ws;     ws += (size_t)B_DIM * OUT_DIM * 4;
    float*    q      = (float*)ws;     ws += (size_t)B_DIM * OUT_DIM * 4;

    // 1) sigmoid transfer function + bf16 packing of s, s^2
    k_prep_s<<<(OUT_DIM * IN_DIM / 2) / 256, 256, 0, stream>>>(
        weight, x0, dx, a, dpar, s_f32, s_bf, s2_bf);

    // 2) bf16 packing of input, input^2
    k_prep_in<<<(B_DIM * IN_DIM / 2) / 256, 256, 0, stream>>>(
        input, in_bf, in2_bf);

    // 3) WMMA GEMMs: mu = in . s^T, q = in^2 . (s^2)^T
    //    512 tiles of 16x16, 4 waves/block -> 128 blocks
    k_wmma_mu_q<<<128, 128, 0, stream>>>(in_bf, in2_bf, s_bf, s2_bf, mu, q);

    // 4) Bernoulli sampling + combine: 512 blocks of 256 threads
    k_bern_combine<<<(B_DIM / 16) * (OUT_DIM / 16), 256, 0, stream>>>(
        input, s_f32, mu, q, bias, out);
}
